// GroupedQuerySelfAttention_36799279792938
// MI455X (gfx1250) — matrix operand order repaired and checked by reading, not compiled, not run on hardware
//
#include <hip/hip_runtime.h>
#include <hip/hip_bf16.h>

typedef __attribute__((ext_vector_type(16))) __bf16 v16bf;
typedef __attribute__((ext_vector_type(8)))  float  v8f;

namespace {
constexpr int BATCH = 2;
constexpr int T     = 2048;
constexpr int DM    = 2048;
constexpr int HQ    = 32;
constexpr int HKV   = 8;
constexpr int HD    = 64;
constexpr int BT    = BATCH * T;            // 4096 rows
constexpr int NQKV  = (HQ + 2 * HKV) * HD;  // 3072 fused QKV outputs
}

union Frag {
  unsigned u[8];
  v16bf    v;
};

__device__ __forceinline__ v8f v8f_zero() {
  v8f z = {0.f, 0.f, 0.f, 0.f, 0.f, 0.f, 0.f, 0.f};
  return z;
}

__device__ __forceinline__ unsigned short f2bf(float f) {
  unsigned x = __builtin_bit_cast(unsigned, f);
  x += 0x7FFFu + ((x >> 16) & 1u);   // round-to-nearest-even
  return (unsigned short)(x >> 16);
}

__device__ __forceinline__ v8f wmma_bf16(const Frag& a, const Frag& b, v8f c) {
  // (neg_a, A, neg_b, B, c_mod, C, reuse_a, reuse_b)
  return __builtin_amdgcn_wmma_f32_16x16x32_bf16(false, a.v, false, b.v,
                                                 (short)0, c, false, false);
}

// ---------------------------------------------------------------- cast ----
__global__ __launch_bounds__(256) void cast_f32_bf16(
    const float* __restrict__ src, unsigned short* __restrict__ dst, int n) {
  int i = blockIdx.x * blockDim.x + threadIdx.x;
  if (i < n) dst[i] = f2bf(src[i]);
}

// ------------------------------------------------- C = A(MxK) * W(NxK)^T --
// bf16 inputs, fp32 out. Block = 8 waves (2x4); each wave owns a 32x64 tile
// (block tile 64x256). 12 b128 loads feed 8 WMMAs per 32-wide K chunk.
__global__ __launch_bounds__(256) void gemm_bf16(
    const unsigned short* __restrict__ A, const unsigned short* __restrict__ Bw,
    float* __restrict__ C, int M, int N, int K) {
  const int lane = threadIdx.x & 31;
  const int wave = threadIdx.x >> 5;
  const int l16  = lane & 15;
  const int half = lane >> 4;
  const int m0 = blockIdx.y * 64 + (wave >> 2) * 32;
  const int n0 = blockIdx.x * 256 + (wave & 3) * 64;

  const unsigned* Ar[2];
  const unsigned* Br[4];
#pragma unroll
  for (int mi = 0; mi < 2; ++mi)
    Ar[mi] = (const unsigned*)(A + (size_t)(m0 + mi * 16 + l16) * K);
#pragma unroll
  for (int ni = 0; ni < 4; ++ni)
    Br[ni] = (const unsigned*)(Bw + (size_t)(n0 + ni * 16 + l16) * K);

  v8f c[2][4];
#pragma unroll
  for (int mi = 0; mi < 2; ++mi)
#pragma unroll
    for (int ni = 0; ni < 4; ++ni) c[mi][ni] = v8f_zero();

  const int awoff = half << 2;  // + ((v&4)<<1) + (v&3)  (dword units)
  const int bwoff = half << 2;  // + ((v&4)<<1) + (v&3)

  for (int k0 = 0; k0 < K; k0 += 32) {
    const int kw = k0 >> 1;
    __builtin_prefetch(Br[0] + kw + 16, 0, 1);
    Frag a[2], b[4];
#pragma unroll
    for (int v = 0; v < 8; ++v) {
      const int ao = kw + awoff + ((v & 4) << 1) + (v & 3);
      const int bo = kw + bwoff + ((v & 4) << 1) + (v & 3);
#pragma unroll
      for (int mi = 0; mi < 2; ++mi) a[mi].u[v] = Ar[mi][ao];
#pragma unroll
      for (int ni = 0; ni < 4; ++ni) b[ni].u[v] = Br[ni][bo];
    }
#pragma unroll
    for (int mi = 0; mi < 2; ++mi)
#pragma unroll
      for (int ni = 0; ni < 4; ++ni)
        c[mi][ni] = wmma_bf16(a[mi], b[ni], c[mi][ni]);
  }

#pragma unroll
  for (int mi = 0; mi < 2; ++mi)
#pragma unroll
    for (int ni = 0; ni < 4; ++ni)
#pragma unroll
      for (int v = 0; v < 8; ++v)
        C[(size_t)(m0 + mi * 16 + v + 8 * half) * N + (n0 + ni * 16 + l16)] =
            c[mi][ni][v];
}

// ------------------------------------- RoPE + head-major repack to bf16 ---
__global__ __launch_bounds__(256) void rope_pack(
    const float* __restrict__ qkv, const int* __restrict__ pos,
    unsigned short* __restrict__ Qb, unsigned short* __restrict__ Kb,
    unsigned short* __restrict__ Vb) {
  const long idx = (long)blockIdx.x * blockDim.x + threadIdx.x;
  const long total = (long)BT * (NQKV / 2);
  if (idx >= total) return;
  const int  c  = (int)(idx % (NQKV / 2));
  const long bt = idx / (NQKV / 2);
  const int  t  = (int)(bt % T);
  const int  b  = (int)(bt / T);
  const float* src = qkv + bt * NQKV;
  const float  p   = (float)pos[t];

  if (c < HQ * 32) {  // ---- Q: rope + fold 1/sqrt(HD)
    const int hh = c >> 5, i = c & 31;
    const float x1 = src[hh * HD + 2 * i];
    const float x2 = src[hh * HD + 2 * i + 1];
    const float freq = __expf(-(2.f * i / (float)HD) * 9.2103403719761836f);
    float sn, cs;
    __sincosf(p * freq, &sn, &cs);
    unsigned short* d = Qb + (((size_t)(b * HQ + hh) * T + t) * HD + 2 * i);
    d[0] = f2bf((x1 * cs - x2 * sn) * 0.125f);
    d[1] = f2bf((x1 * sn + x2 * cs) * 0.125f);
  } else if (c < (HQ + HKV) * 32) {  // ---- K: rope
    const int cc = c - HQ * 32;
    const int hh = cc >> 5, i = cc & 31;
    const float x1 = src[HQ * HD + hh * HD + 2 * i];
    const float x2 = src[HQ * HD + hh * HD + 2 * i + 1];
    const float freq = __expf(-(2.f * i / (float)HD) * 9.2103403719761836f);
    float sn, cs;
    __sincosf(p * freq, &sn, &cs);
    unsigned short* d = Kb + (((size_t)(b * HKV + hh) * T + t) * HD + 2 * i);
    d[0] = f2bf(x1 * cs - x2 * sn);
    d[1] = f2bf(x1 * sn + x2 * cs);
  } else {  // ---- V: plain cast
    const int cc = c - (HQ + HKV) * 32;
    const int hh = cc >> 5, i = cc & 31;
    unsigned short* d = Vb + (((size_t)(b * HKV + hh) * T + t) * HD + 2 * i);
    d[0] = f2bf(src[(HQ + HKV) * HD + hh * HD + 2 * i]);
    d[1] = f2bf(src[(HQ + HKV) * HD + hh * HD + 2 * i + 1]);
  }
}

// --------------------------------------------- causal flash attention -----
// Block: 128 threads = 4 waves; each wave owns a 16-row Q strip of a 64-row
// tile for one (b, q-head). K tiles of 64 keys; V staged TRANSPOSED in LDS
// so PV B-fragments are single dword reads; P goes through LDS to convert
// C-layout -> A-layout for the PV WMMA.
__global__ __launch_bounds__(128) void flash_attn(
    const unsigned short* __restrict__ Q, const unsigned short* __restrict__ Kg,
    const unsigned short* __restrict__ Vg, unsigned short* __restrict__ O) {
  const int qt  = blockIdx.x;
  const int h   = blockIdx.y;
  const int b   = blockIdx.z;
  const int kvh = h >> 2;  // G = 4
  const int lane = threadIdx.x & 31;
  const int wave = threadIdx.x >> 5;
  const int l16  = lane & 15;
  const int half = lane >> 4;
  const int qrow0 = qt * 64 + wave * 16;

  __shared__ unsigned short VtT[64][72];     // [d][key], padded stride
  __shared__ unsigned short Pt[4][16][72];   // per-wave P scratch

  const unsigned short* Qp = Q + (size_t)(b * HQ + h) * T * HD;
  const unsigned short* Kp = Kg + (size_t)(b * HKV + kvh) * T * HD;
  const unsigned short* Vp = Vg + (size_t)(b * HKV + kvh) * T * HD;

  // Q fragments (16 rows x 64, two K-chunks of 32) pinned in registers.
  Frag qf[2];
  {
    const unsigned* Qrow = (const unsigned*)(Qp + (size_t)(qrow0 + l16) * HD);
#pragma unroll
    for (int ch = 0; ch < 2; ++ch)
#pragma unroll
      for (int v = 0; v < 8; ++v)
        qf[ch].u[v] = Qrow[ch * 16 + (half << 2) + ((v & 4) << 1) + (v & 3)];
  }

  v8f o[4];
  o[0] = v8f_zero(); o[1] = v8f_zero(); o[2] = v8f_zero(); o[3] = v8f_zero();
  float m[8], l[8];
#pragma unroll
  for (int v = 0; v < 8; ++v) { m[v] = -3.0e38f; l[v] = 0.f; }

  for (int j = 0; j <= qt; ++j) {
    const int kb = j * 64;

    {  // stage V tile transposed: coalesced global b128 reads, b16 scatter
      const int r  = threadIdx.x >> 1;           // key row 0..63
      const int c0 = (threadIdx.x & 1) * 32;     // d base
      const unsigned* src = (const unsigned*)(Vp + (size_t)(kb + r) * HD + c0);
#pragma unroll
      for (int i = 0; i < 16; ++i) {
        const unsigned w = src[i];
        const int d0 = c0 + 2 * i;
        VtT[d0][r]     = (unsigned short)(w & 0xFFFFu);
        VtT[d0 + 1][r] = (unsigned short)(w >> 16);
      }
    }
    __syncthreads();

    // S = Q * K^T  (K fragments contiguous from global)
    float s[4][8];
#pragma unroll
    for (int nt = 0; nt < 4; ++nt) {
      v8f acc = v8f_zero();
      const unsigned* Krow =
          (const unsigned*)(Kp + (size_t)(kb + nt * 16 + l16) * HD);
#pragma unroll
      for (int ch = 0; ch < 2; ++ch) {
        Frag kf;
#pragma unroll
        for (int v = 0; v < 8; ++v)
          kf.u[v] = Krow[ch * 16 + (half << 2) + ((v & 4) << 1) + (v & 3)];
        acc = wmma_bf16(qf[ch], kf, acc);
      }
      const int col = kb + nt * 16 + l16;
#pragma unroll
      for (int v = 0; v < 8; ++v) {
        const int row = qrow0 + v + 8 * half;
        s[nt][v] = (col <= row) ? acc[v] : -3.0e38f;
      }
    }

    // online softmax: row max via intra-16-lane shuffles
    float alpha[8];
#pragma unroll
    for (int v = 0; v < 8; ++v) {
      float mt = fmaxf(fmaxf(s[0][v], s[1][v]), fmaxf(s[2][v], s[3][v]));
      mt = fmaxf(mt, __shfl_xor(mt, 1));
      mt = fmaxf(mt, __shfl_xor(mt, 2));
      mt = fmaxf(mt, __shfl_xor(mt, 4));
      mt = fmaxf(mt, __shfl_xor(mt, 8));
      const float mn = fmaxf(m[v], mt);
      alpha[v] = __expf(m[v] - mn);
      m[v] = mn;
    }

#pragma unroll
    for (int v = 0; v < 8; ++v) {
      float rs = 0.f;
#pragma unroll
      for (int nt = 0; nt < 4; ++nt) {
        const float p = __expf(s[nt][v] - m[v]);
        rs += p;
        Pt[wave][v + 8 * half][nt * 16 + l16] = f2bf(p);
      }
      rs += __shfl_xor(rs, 1);
      rs += __shfl_xor(rs, 2);
      rs += __shfl_xor(rs, 4);
      rs += __shfl_xor(rs, 8);
      l[v] = l[v] * alpha[v] + rs;
    }

#pragma unroll
    for (int nt = 0; nt < 4; ++nt)
#pragma unroll
      for (int v = 0; v < 8; ++v) o[nt][v] *= alpha[v];

    // P in A-layout from LDS (same-wave LDS ops are in order)
    Frag pa[2];
    {
      const unsigned* Prow = (const unsigned*)&Pt[wave][l16][0];
#pragma unroll
      for (int ch = 0; ch < 2; ++ch)
#pragma unroll
        for (int v = 0; v < 8; ++v)
          pa[ch].u[v] = Prow[ch * 16 + (half << 2) + ((v & 4) << 1) + (v & 3)];
    }

    // O += P * V : V B-fragment = single dword per element from VtT
#pragma unroll
    for (int nt = 0; nt < 4; ++nt) {
      const unsigned* Vrow = (const unsigned*)&VtT[nt * 16 + l16][0];
#pragma unroll
      for (int ch = 0; ch < 2; ++ch) {
        Frag vf;
#pragma unroll
        for (int v = 0; v < 8; ++v)
          vf.u[v] = Vrow[ch * 16 + (half << 2) + ((v & 4) << 1) + (v & 3)];
        o[nt] = wmma_bf16(pa[ch], vf, o[nt]);
      }
    }
    __syncthreads();  // before VtT is overwritten next iteration
  }

#pragma unroll
  for (int nt = 0; nt < 4; ++nt)
#pragma unroll
    for (int v = 0; v < 8; ++v) {
      const int row = qrow0 + v + 8 * half;
      const int d   = nt * 16 + l16;
      O[((size_t)b * T + row) * (HQ * HD) + h * HD + d] =
          f2bf(o[nt][v] * (1.f / l[v]));
    }
}

// --------------------------------------------------------------- driver ---
extern "C" void kernel_launch(void* const* d_in, const int* in_sizes, int n_in,
                              void* d_out, int out_size, void* d_ws,
                              size_t ws_size, hipStream_t stream) {
  (void)in_sizes; (void)n_in; (void)out_size; (void)ws_size;
  const float* x   = (const float*)d_in[0];
  const int*   pos = (const int*)d_in[1];
  const float* WQ  = (const float*)d_in[2];
  const float* WK  = (const float*)d_in[3];
  const float* WV  = (const float*)d_in[4];
  const float* WO  = (const float*)d_in[5];
  float* out = (float*)d_out;

  char* ws = (char*)d_ws;
  unsigned short* xb   = (unsigned short*)ws; ws += (size_t)BT * DM * 2;
  unsigned short* Wqkv = (unsigned short*)ws; ws += (size_t)NQKV * DM * 2;
  unsigned short* WOb  = (unsigned short*)ws; ws += (size_t)DM * DM * 2;
  float*          qkv  = (float*)ws;          ws += (size_t)BT * NQKV * 4;
  unsigned short* Qb   = (unsigned short*)ws; ws += (size_t)BATCH * HQ  * T * HD * 2;
  unsigned short* Kb   = (unsigned short*)ws; ws += (size_t)BATCH * HKV * T * HD * 2;
  unsigned short* Vb   = (unsigned short*)ws; ws += (size_t)BATCH * HKV * T * HD * 2;
  unsigned short* Oatt = (unsigned short*)ws;

  const int nX  = BT * DM;
  const int nWq = HQ * HD * DM;
  const int nWk = HKV * HD * DM;
  const int nWo = DM * DM;
  cast_f32_bf16<<<(nX  + 255) / 256, 256, 0, stream>>>(x,  xb,   nX);
  cast_f32_bf16<<<(nWq + 255) / 256, 256, 0, stream>>>(WQ, Wqkv, nWq);
  cast_f32_bf16<<<(nWk + 255) / 256, 256, 0, stream>>>(WK, Wqkv + (size_t)nWq, nWk);
  cast_f32_bf16<<<(nWk + 255) / 256, 256, 0, stream>>>(WV, Wqkv + (size_t)nWq + nWk, nWk);
  cast_f32_bf16<<<(nWo + 255) / 256, 256, 0, stream>>>(WO, WOb, nWo);

  // Fused QKV projection: (4096 x 2048) @ (3072 x 2048)^T
  gemm_bf16<<<dim3(NQKV / 256, BT / 64), 256, 0, stream>>>(
      xb, Wqkv, qkv, BT, NQKV, DM);

  // RoPE + repack to (B, H, T, HD) bf16
  const long pairs = (long)BT * (NQKV / 2);
  rope_pack<<<(unsigned)((pairs + 255) / 256), 256, 0, stream>>>(
      qkv, pos, Qb, Kb, Vb);

  // Causal GQA flash attention
  flash_attn<<<dim3(T / 64, HQ, BATCH), 128, 0, stream>>>(Qb, Kb, Vb, Oatt);

  // Output projection: (4096 x 2048) @ (2048 x 2048)^T -> fp32 out
  gemm_bf16<<<dim3(DM / 256, BT / 64), 256, 0, stream>>>(
      Oatt, WOb, out, BT, DM, DM);
}
